// RNNBio_19619410608758
// MI455X (gfx1250) — compile-verified, matrix operand order checked by reading, not run on hardware
//
#include <hip/hip_runtime.h>

typedef __attribute__((ext_vector_type(16))) __bf16 v16bf;
typedef __attribute__((ext_vector_type(8)))  float  v8f;

#define T_STEPS 512
#define BATCH    64
#define INPUT   512
#define HIDDEN 1024

static __device__ __forceinline__ unsigned short f32_to_bf16(float f) {
  unsigned int u = __builtin_bit_cast(unsigned int, f);
  unsigned int r = u + 0x7FFFu + ((u >> 16) & 1u);   // round-to-nearest-even
  return (unsigned short)(r >> 16);
}

union V16BF { v16bf v; uint4 q[2]; };

// ---------------------------------------------------------------------------
// Kernel 0: one-time f32 -> bf16 conversion of Whh and Wxh into workspace.
// ---------------------------------------------------------------------------
__global__ __launch_bounds__(256) void convert_weights(
    const float* __restrict__ Whh, const float* __restrict__ Wxh,
    unsigned short* __restrict__ whh_bf, unsigned short* __restrict__ wxh_bf) {
  int i = blockIdx.x * blockDim.x + threadIdx.x;
  const int NW = HIDDEN * HIDDEN;
  if (i < NW) {
    whh_bf[i] = f32_to_bf16(Whh[i]);
  } else {
    int j = i - NW;                       // grid sized exactly, no OOB
    wxh_bf[j] = f32_to_bf16(Wxh[j]);
  }
}

// ---------------------------------------------------------------------------
// Kernel 1: pre[t,b,n] = X[t,b,:] . Wxh[n,:] + bh[n]   (bulk parallel GEMM)
//   A = X flattened (T*B, INPUT) row-major, B[k][n] = Wxh[n][k] (K-contiguous)
//   Workgroup: 128 threads = 4 waves; 16-row M tile staged as bf16 in LDS;
//   each wave owns 2 N-tiles of 16 cols.
// ---------------------------------------------------------------------------
__global__ __launch_bounds__(128) void input_gemm(
    const float* __restrict__ X,
    const unsigned short* __restrict__ wxh_bf,
    const float* __restrict__ bh,
    float* __restrict__ out) {
  __shared__ __align__(16) unsigned short Alds[16 * INPUT];   // 16 KB
  const int tid   = threadIdx.x;
  const int mtile = blockIdx.x >> 3;          // 0 .. 2047
  const int ng    = blockIdx.x & 7;           // 0 .. 7  (128 cols each)
  const int m0    = mtile * 16;

  for (int i = tid; i < 16 * INPUT; i += 128) {
    int row = i >> 9, k = i & (INPUT - 1);
    Alds[i] = f32_to_bf16(X[(size_t)(m0 + row) * INPUT + k]);
  }
  __syncthreads();

  const int lane = tid & 31, w = tid >> 5;
  const int cit  = lane & 15, half = lane >> 4;
  const int n0   = ng * 128 + w * 32;

  v8f acc0 = {}; v8f acc1 = {};
  const unsigned short* ap  = Alds + cit * INPUT + 8 * half;
  const unsigned short* b0p = wxh_bf + (size_t)(n0 + cit)      * INPUT + 8 * half;
  const unsigned short* b1p = wxh_bf + (size_t)(n0 + 16 + cit) * INPUT + 8 * half;

  for (int kk = 0; kk < INPUT; kk += 32) {
    V16BF a, b0, b1;
    a.q[0]  = *(const uint4*)(ap  + kk);
    a.q[1]  = *(const uint4*)(ap  + kk + 16);
    b0.q[0] = *(const uint4*)(b0p + kk);
    b0.q[1] = *(const uint4*)(b0p + kk + 16);
    b1.q[0] = *(const uint4*)(b1p + kk);
    b1.q[1] = *(const uint4*)(b1p + kk + 16);
    acc0 = __builtin_amdgcn_wmma_f32_16x16x32_bf16(false, a.v, false, b0.v,
                                                   (short)0, acc0, false, false);
    acc1 = __builtin_amdgcn_wmma_f32_16x16x32_bf16(false, a.v, false, b1.v,
                                                   (short)0, acc1, false, false);
  }

  for (int j = 0; j < 2; ++j) {
    v8f acc = j ? acc1 : acc0;
    int col = n0 + j * 16 + cit;
    float bias = bh[col];
    for (int r = 0; r < 8; ++r) {
      int row = m0 + r + 8 * half;
      out[(size_t)row * HIDDEN + col] = acc[r] + bias;
    }
  }
}

// ---------------------------------------------------------------------------
// Kernel 2: one recurrent step.
//   cur holds pre-activation (x_t@Wxh^T + bh); prev holds h[t-1] (or null at
//   t=0).  Compute rec = relu(h_prev @ Whh^T) * ei, then
//   cur <- tanh(cur + rec)  in place.
//   B[k][n] = Whh[n][k]  -> K-contiguous rows of the bf16 weight copy (L2
//   resident: 2 MB reused by all 512 steps).
// ---------------------------------------------------------------------------
__global__ __launch_bounds__(128) void rnn_step(
    const float* __restrict__ prev,
    float* __restrict__ cur,
    const unsigned short* __restrict__ whh_bf,
    const float* __restrict__ ei) {
  __shared__ __align__(16) unsigned short Alds[16 * HIDDEN];  // 32 KB
  const int tid   = threadIdx.x;
  const int mtile = blockIdx.x >> 3;          // 0 .. 3   (batch tiles)
  const int ng    = blockIdx.x & 7;           // 0 .. 7   (128 cols each)
  const int m0    = mtile * 16;

  for (int i = tid; i < 16 * HIDDEN; i += 128) {
    int row = i >> 10, k = i & (HIDDEN - 1);
    float f = prev ? prev[(size_t)(m0 + row) * HIDDEN + k] : 0.0f;
    Alds[i] = f32_to_bf16(f);
  }
  __syncthreads();

  const int lane = tid & 31, w = tid >> 5;
  const int cit  = lane & 15, half = lane >> 4;
  const int n0   = ng * 128 + w * 32;

  v8f acc0 = {}; v8f acc1 = {};
  const unsigned short* ap  = Alds + cit * HIDDEN + 8 * half;
  const unsigned short* b0p = whh_bf + (size_t)(n0 + cit)      * HIDDEN + 8 * half;
  const unsigned short* b1p = whh_bf + (size_t)(n0 + 16 + cit) * HIDDEN + 8 * half;

  for (int kk = 0; kk < HIDDEN; kk += 32) {
    V16BF a, b0, b1;
    a.q[0]  = *(const uint4*)(ap  + kk);
    a.q[1]  = *(const uint4*)(ap  + kk + 16);
    b0.q[0] = *(const uint4*)(b0p + kk);
    b0.q[1] = *(const uint4*)(b0p + kk + 16);
    b1.q[0] = *(const uint4*)(b1p + kk);
    b1.q[1] = *(const uint4*)(b1p + kk + 16);
    acc0 = __builtin_amdgcn_wmma_f32_16x16x32_bf16(false, a.v, false, b0.v,
                                                   (short)0, acc0, false, false);
    acc1 = __builtin_amdgcn_wmma_f32_16x16x32_bf16(false, a.v, false, b1.v,
                                                   (short)0, acc1, false, false);
  }

  for (int j = 0; j < 2; ++j) {
    v8f acc = j ? acc1 : acc0;
    int col = n0 + j * 16 + cit;
    float e = ei[col];
    for (int r = 0; r < 8; ++r) {
      int row = m0 + r + 8 * half;
      size_t idx = (size_t)row * HIDDEN + col;
      float rec = fmaxf(acc[r], 0.0f) * e;
      cur[idx] = tanhf(cur[idx] + rec);
    }
  }
}

// ---------------------------------------------------------------------------
extern "C" void kernel_launch(void* const* d_in, const int* in_sizes, int n_in,
                              void* d_out, int out_size, void* d_ws, size_t ws_size,
                              hipStream_t stream) {
  const float* X   = (const float*)d_in[0];   // (T, B, I)
  const float* Wxh = (const float*)d_in[1];   // (H, I)
  const float* Whh = (const float*)d_in[2];   // (H, H)
  const float* bh  = (const float*)d_in[3];   // (H)
  const float* ei  = (const float*)d_in[4];   // (H)
  float* out = (float*)d_out;                 // (T,B,H) then (B,H) h_last

  unsigned short* whh_bf = (unsigned short*)d_ws;            // 2 MB
  unsigned short* wxh_bf = whh_bf + (size_t)HIDDEN * HIDDEN; // 1 MB

  // 0) weights -> bf16 (exactly covers HIDDEN*HIDDEN + HIDDEN*INPUT elems)
  convert_weights<<<(HIDDEN * HIDDEN + HIDDEN * INPUT) / 256, 256, 0, stream>>>(
      Whh, Wxh, whh_bf, wxh_bf);

  // 1) bulk input projection: out[t] = X[t]@Wxh^T + bh for all t
  input_gemm<<<(T_STEPS * BATCH / 16) * 8, 128, 0, stream>>>(X, wxh_bf, bh, out);

  // 2) sequential recurrence, in place over out[t]
  for (int t = 0; t < T_STEPS; ++t) {
    const float* prev = t ? out + (size_t)(t - 1) * BATCH * HIDDEN : nullptr;
    float* cur = out + (size_t)t * BATCH * HIDDEN;
    rnn_step<<<32, 128, 0, stream>>>(prev, cur, whh_bf, ei);
  }

  // 3) h_last = out[T-1]
  hipMemcpyAsync(out + (size_t)T_STEPS * BATCH * HIDDEN,
                 out + (size_t)(T_STEPS - 1) * BATCH * HIDDEN,
                 (size_t)BATCH * HIDDEN * sizeof(float),
                 hipMemcpyDeviceToDevice, stream);
}
